// MyModel_87522843560192
// MI455X (gfx1250) — matrix operand order repaired and checked by reading, not compiled, not run on hardware
//
#include <hip/hip_runtime.h>
#include <math.h>

typedef __attribute__((ext_vector_type(16))) _Float16 v16h;
typedef __attribute__((ext_vector_type(8)))  float    v8f;

// Problem constants (reference: B=8, T=16, C=8, H=W=128, F=32, 3x3 kernels)
#define BQ   8
#define TT   16
#define CC   8
#define HH   128
#define WW   128
#define FF   32
#define HO   126
#define WO   126
#define K1   72      // C*9   (x-conv part of K)
#define KTOT 360     // 72 + F*9
#define KPAD 384     // padded to 12 chunks of 32
#define NCHUNK 12
#define OC   128     // 4*F gate channels
#define NPX  32      // pixels per block (2 x 16-wide WMMA N tiles)
#define APACK_ELEMS (OC * NCHUNK * 2 * 16)   // 49152 halves = 96 KB

__device__ __forceinline__ float hsig(float x) {
    return fminf(fmaxf(0.2f * x + 0.5f, 0.0f), 1.0f);
}

// ---------------------------------------------------------------------------
// One-time weight pack: f32 Wk/Uk -> f16, laid out per-fragment so each lane's
// 16 A-halves for one WMMA chunk are one contiguous 32B load.
//   Fragment element j (contiguous k order, lane-half `hi`, matches B):
//     j in [0,16): K = kc*32 + hi*16 + j
//   K order: [0,72) = Wk[oc][..], [72,360) = Uk[oc][..], [360,384) = 0 pad.
// ---------------------------------------------------------------------------
__global__ __launch_bounds__(256) void pack_weights(
    const float* __restrict__ Wk, const float* __restrict__ Uk,
    _Float16* __restrict__ Ap)
{
    int idx = blockIdx.x * 256 + threadIdx.x;
    if (idx >= APACK_ELEMS) return;
    int j  = idx & 15;
    int hi = (idx >> 4) & 1;
    int kc = (idx >> 5) % NCHUNK;
    int oc = idx / (NCHUNK * 2 * 16);
    int k  = kc * 32 + hi * 16 + j;
    float v = 0.0f;
    if (k < K1)        v = Wk[oc * K1 + k];
    else if (k < KTOT) v = Uk[oc * (KTOT - K1) + (k - K1)];
    Ap[idx] = (_Float16)v;
}

// ---------------------------------------------------------------------------
// One ConvLSTM timestep: implicit GEMM z = [Wk|Uk] x im2col([x_t ; h]) via
// v_wmma_f32_16x16x32_f16, fused with gate math and c/h update.
// Each block: 128 OC x 32 px.  Each wave: 16 OC x 32 px (2 acc tiles,
// A fragment reused across both N tiles).
// ---------------------------------------------------------------------------
__global__ __launch_bounds__(256) void convlstm_step(
    const float*    __restrict__ x,     // (B,T,C,H,W)
    const _Float16* __restrict__ Ap,    // packed weights (fragment order)
    const float*    __restrict__ bias,  // (128,)
    const float*    __restrict__ h_in,  // (B,32,126,126)
    float*          __restrict__ cbuf,  // (B,32,126,126) in-place
    float*          __restrict__ h_out, // (B,32,126,126)
    int t)
{
    __shared__ _Float16 Blds[NPX * KPAD];   // im2col panel [px][k], 24 KB
    __shared__ float    zbuf[OC * (NPX + 1)]; // z tile [oc][px], stride-33

    const int bat  = blockIdx.z;
    const int row  = blockIdx.y;           // output row 0..125
    const int col0 = blockIdx.x * NPX;     // output col base

    const float* xt = x + ((size_t)(bat * TT + t) * CC) * HH * WW;
    const float* hb = h_in + (size_t)bat * FF * HO * WO;

    // ---- Build im2col B panel (x part: VALID; h part: SAME, zero-padded) ----
    for (int idx = threadIdx.x; idx < NPX * KPAD; idx += 256) {
        int px = idx & (NPX - 1);
        int k  = idx >> 5;
        float v = 0.0f;
        if (k < K1) {
            int c = k / 9, rem = k - c * 9, r = rem / 3, s = rem - r * 3;
            int ri = row + r;            // 0..127, always in range
            int cl = col0 + px + s;      // exceeds 127 only for dead px
            if (cl < WW) v = xt[((size_t)c * HH + ri) * WW + cl];
        } else if (k < KTOT) {
            int kk = k - K1;
            int f = kk / 9, rem = kk - f * 9, r = rem / 3, s = rem - r * 3;
            int ri = row - 1 + r;
            int cl = col0 + px - 1 + s;
            if (ri >= 0 && ri < HO && cl >= 0 && cl < WO)
                v = hb[((size_t)f * HO + ri) * WO + cl];
        }
        Blds[px * KPAD + k] = (_Float16)v;
    }
    __syncthreads();

    // ---- Implicit GEMM: each wave computes a 16-OC x 32-px tile ----
    const int lane   = threadIdx.x & 31;
    const int wid    = threadIdx.x >> 5;
    const int ocbase = wid * 16;
    const int m      = lane & 15;
    const int hiL    = (lane >> 4) & 1;

    // Bias folded into accumulators (C/D layout: lane-hi half -> M = v+8)
    v8f acc0, acc1;
    #pragma unroll
    for (int v = 0; v < 8; ++v) acc0[v] = bias[ocbase + v + hiL * 8];
    acc1 = acc0;

    const _Float16* arow  = Ap + ((size_t)((ocbase + m) * NCHUNK) * 2 + hiL) * 16;
    const _Float16* brow0 = &Blds[m * KPAD];          // px group 0..15
    const _Float16* brow1 = &Blds[(16 + m) * KPAD];   // px group 16..31
    #pragma unroll
    for (int kc = 0; kc < NCHUNK; ++kc) {
        v16h afrag  = *(const v16h*)(arow + kc * 32);           // 32B packed load
        v16h bfrag0 = *(const v16h*)(brow0 + kc * 32 + hiL * 16);
        v16h bfrag1 = *(const v16h*)(brow1 + kc * 32 + hiL * 16);
        acc0 = __builtin_amdgcn_wmma_f32_16x16x32_f16(
            false, afrag, false, bfrag0, (short)0, acc0, false, false);
        acc1 = __builtin_amdgcn_wmma_f32_16x16x32_f16(
            false, afrag, false, bfrag1, (short)0, acc1, false, false);
    }

    // ---- Stage z tiles to LDS ----
    #pragma unroll
    for (int v = 0; v < 8; ++v) {
        zbuf[(ocbase + v + hiL * 8) * (NPX + 1) + m]      = acc0[v];
        zbuf[(ocbase + v + hiL * 8) * (NPX + 1) + 16 + m] = acc1[v];
    }
    __syncthreads();

    // ---- Fused gate math: i,f,g,o -> c,h update ----
    for (int e = threadIdx.x; e < FF * NPX; e += 256) {
        int px  = e & (NPX - 1);
        int fch = e >> 5;
        int col = col0 + px;
        if (col < WO) {
            float zi = zbuf[(      fch) * (NPX + 1) + px];
            float zf = zbuf[( 32 + fch) * (NPX + 1) + px];
            float zg = zbuf[( 64 + fch) * (NPX + 1) + px];
            float zo = zbuf[( 96 + fch) * (NPX + 1) + px];
            size_t idx = (((size_t)bat * FF + fch) * HO + row) * WO + col;
            float cp = cbuf[idx];
            float cn = hsig(zf) * cp + hsig(zi) * tanhf(zg);
            float hn = hsig(zo) * tanhf(cn);
            cbuf[idx]  = cn;
            h_out[idx] = hn;
        }
    }
}

extern "C" void kernel_launch(void* const* d_in, const int* in_sizes, int n_in,
                              void* d_out, int out_size, void* d_ws, size_t ws_size,
                              hipStream_t stream) {
    const float* x  = (const float*)d_in[0];
    const float* Wk = (const float*)d_in[1];
    const float* Uk = (const float*)d_in[2];
    const float* b  = (const float*)d_in[3];
    float* out = (float*)d_out;

    const size_t NH = (size_t)BQ * FF * HO * WO;   // 4,064,256 elems
    float*     h0 = (float*)d_ws;
    float*     h1 = h0 + NH;
    float*     cb = h1 + NH;
    _Float16*  Ap = (_Float16*)(cb + NH);          // 96 KB packed weights

    // Zero h0 (step-0 hidden) and c; h1/Ap are fully written before first read.
    hipMemsetAsync(d_ws, 0, 3 * NH * sizeof(float), stream);

    pack_weights<<<(APACK_ELEMS + 255) / 256, 256, 0, stream>>>(Wk, Uk, Ap);

    dim3 grid((WO + NPX - 1) / NPX, HO, BQ);   // (4, 126, 8)
    for (int t = 0; t < TT; ++t) {
        const float* hin = (t & 1) ? h1 : h0;
        float* hout = (t == TT - 1) ? out : ((t & 1) ? h0 : h1);
        convlstm_step<<<grid, 256, 0, stream>>>(x, Ap, b, hin, cb, hout, t);
    }
}
